// SimpleResRNNTbptt_89756226552407
// MI455X (gfx1250) — matrix operand order repaired and checked by reading, not compiled, not run on hardware
//
#include <hip/hip_runtime.h>
#include <hip/hip_bf16.h>

// ---------------- problem constants ----------------
#define T_STEPS 1000
#define BATCH   256
#define INF_    700          // input features
#define HID     256
#define OUTC    20
#define KPAD    704          // 700 padded to multiple of 32

#define DELTA_  0.01f
#define GAMMA_  0.9f
#define THETA_  1.0f

// ---------------- d_out layout (floats) ----------------
static constexpr size_t OUTPUTS_OFF = 0;                                   // [T,B,O]
static constexpr size_t LOSS_OFF    = (size_t)T_STEPS * BATCH * OUTC;      // 5,120,000
static constexpr size_t ZF_OFF      = LOSS_OFF + 1;
static constexpr size_t UF_OFF      = ZF_OFF + (size_t)BATCH * HID;
static constexpr size_t OUF_OFF     = UF_OFF + (size_t)BATCH * HID;

// ---------------- workspace layout (bytes) ----------------
static constexpr size_t WS_WX   = 0;                       // bf16 [256][704]
static constexpr size_t WS_WZ   = 360448;                  // bf16 [256][256]
static constexpr size_t WS_WO   = WS_WZ + 131072;          // bf16 [32][256] (rows >=20 zero)
static constexpr size_t WS_LOSS = WS_WO + 16384;           // f32  [256]
static constexpr size_t WS_CUR  = 524288;                  // f32  [T*B][256]

typedef __attribute__((ext_vector_type(16))) __bf16 v16bf;
typedef __attribute__((ext_vector_type(8)))  __bf16 v8bf;
typedef __attribute__((ext_vector_type(8)))  float  v8f;

static __device__ __forceinline__ v8f wmma_bf16(v16bf a, v16bf b, v8f c) {
  // D = A(16x32 bf16) * B(32x16 bf16) + C(16x16 f32)
  return __builtin_amdgcn_wmma_f32_16x16x32_bf16(false, a, false, b, (short)0, c,
                                                 false, false);
}

// A fragment: lane L holds row M=L%16; element j -> K = kbase + 16*(j>>3) + 8*(L>>4) + (j&7)
static __device__ __forceinline__ v16bf load_a_frag(const __bf16* row, int kbase, int laneHi) {
  v8bf lo = *(const v8bf*)(row + kbase + 8 * laneHi);
  v8bf hi = *(const v8bf*)(row + kbase + 16 + 8 * laneHi);
  v16bf a;
#pragma unroll
  for (int i = 0; i < 8; ++i) { a[i] = lo[i]; a[i + 8] = hi[i]; }
  return a;
}

// B fragment for B[k][n] = W[n][k]: lane L holds column n; element j -> K = kbase + 16*(j>>3) + 8*(L>>4) + (j&7)
static __device__ __forceinline__ v16bf load_b_frag(const __bf16* mat, int ldk, int n,
                                                    int kbase, int laneHi) {
  const __bf16* p = mat + (size_t)n * ldk + kbase + 8 * laneHi;
  v8bf lo = *(const v8bf*)(p);
  v8bf hi = *(const v8bf*)(p + 16);
  v16bf b;
#pragma unroll
  for (int i = 0; i < 8; ++i) { b[i] = lo[i]; b[i + 8] = hi[i]; }
  return b;
}

static __device__ __forceinline__ unsigned int pack_bf16(float a, float b) {
  unsigned short lo = __builtin_bit_cast(unsigned short, (__bf16)a);
  unsigned short hi = __builtin_bit_cast(unsigned short, (__bf16)b);
  return ((unsigned int)hi << 16) | lo;
}

// ---------------- kernel 0: convert weights to bf16, zero-padded ----------------
__global__ __launch_bounds__(256) void prep_weights_kernel(
    const float* __restrict__ W_h, const float* __restrict__ W_o,
    __bf16* __restrict__ wx, __bf16* __restrict__ wz, __bf16* __restrict__ wo) {
  int idx = blockIdx.x * 256 + threadIdx.x;
  if (idx < 256 * KPAD) {                       // W_x = W_h[:, :700]  -> [256][704]
    int n = idx / KPAD, k = idx % KPAD;
    wx[idx] = (k < INF_) ? (__bf16)W_h[n * (INF_ + HID) + k] : (__bf16)0.f;
  }
  if (idx < 256 * 256) {                        // W_z = W_h[:, 700:]  -> [256][256]
    int n = idx >> 8, k = idx & 255;
    wz[idx] = (__bf16)W_h[n * (INF_ + HID) + INF_ + k];
  }
  if (idx < 32 * 256) {                         // W_o padded to [32][256]
    int n = idx >> 8, k = idx & 255;
    wo[idx] = (n < OUTC) ? (__bf16)W_o[n * HID + k] : (__bf16)0.f;
  }
}

// ---------------- kernel 1: cur_x[t*B+b][h] = x @ W_x^T (bf16 WMMA, f32 acc) ----------------
__global__ __launch_bounds__(512) void gemm_x_kernel(
    const float* __restrict__ x, const __bf16* __restrict__ wx,
    float* __restrict__ cur) {
  __shared__ __bf16 xlds[16 * KPAD];            // 22 KB: one 16-row M-tile of x, bf16
  const int tid = threadIdx.x;
  const int lane = tid & 31, wid = tid >> 5;    // 16 waves, wave = N-tile
  const int laneHi = lane >> 4, laneLo = lane & 15;
  const size_t g0 = (size_t)blockIdx.x * 16;    // flat (t*B+b) row base

  // cooperative f32->bf16 staged load (float2 reads, packed b32 LDS stores, no conflicts)
  {
    const int row = tid >> 5;                   // 0..15
    const int kp = tid & 31;                    // pair index within 32
    const float2* xrow2 = (const float2*)(x + (g0 + row) * INF_);
    unsigned int* xlu = (unsigned int*)xlds;
#pragma unroll
    for (int i = 0; i < 11; ++i) {              // 352 bf16-pairs per row
      int p = kp + 32 * i;
      float2 vv;
      if (p < INF_ / 2) vv = xrow2[p]; else { vv.x = 0.f; vv.y = 0.f; }
      xlu[row * (KPAD / 2) + p] = pack_bf16(vv.x, vv.y);
    }
  }
  __syncthreads();

  const __bf16* arow = xlds + laneLo * KPAD;
  const int n = wid * 16 + laneLo;
  v8f acc = {0.f, 0.f, 0.f, 0.f, 0.f, 0.f, 0.f, 0.f};
#pragma unroll 2
  for (int ks = 0; ks < 22; ++ks) {
    v16bf a = load_a_frag(arow, ks * 32, laneHi);
    v16bf b = load_b_frag(wx, KPAD, n, ks * 32, laneHi);
    acc = wmma_bf16(a, b, acc);
  }
#pragma unroll
  for (int r = 0; r < 8; ++r) {
    int m = r + 8 * laneHi;
    cur[(g0 + m) * HID + n] = acc[r];
  }
}

// ---------------- kernel 2: sequential BRF scan (16 WGs x 16 batch rows) ----------------
__global__ __launch_bounds__(512) void brf_scan_kernel(
    const float* __restrict__ cur, const __bf16* __restrict__ wz,
    const __bf16* __restrict__ wo, const int* __restrict__ y,
    const float* __restrict__ omega, const float* __restrict__ b_offset,
    const float* __restrict__ tau_mem, float* __restrict__ d_out,
    float* __restrict__ loss_part) {
  __shared__ __bf16 zlds[16 * 264];             // z_t tile, bf16, padded rows
  __shared__ float zsl[16 * 32];                // per-step z @ Wo^T staging

  const int tid = threadIdx.x;
  const int lane = tid & 31;
  const int wid = tid >> 5;                     // wave = hidden N-tile; also readout row
  const int laneHi = lane >> 4, laneLo = lane & 15;
  const int n = wid * 16 + laneLo;              // hidden index owned by this lane
  const int b0 = blockIdx.x * 16;               // batch row base

  for (int i = tid; i < 16 * 264; i += 512) zlds[i] = (__bf16)0.f;

  // readout: wave `wid` owns batch row (b0+wid); lane = output channel o
  float alph = 0.f;
  if (lane < OUTC) alph = __expf(-DELTA_ / tau_mem[lane]);
  float ou = 0.f;                               // LI readout state (register)

  const float omv = fabsf(omega[n]);
  const float dom = DELTA_ * omv;
  const float p_om = (-1.f + sqrtf(1.f - dom * dom)) / DELTA_;
  const float b0c = p_om - fabsf(b_offset[n]);

  float u[8], v[8], q[8], z[8], cx[8], cxn[8];
#pragma unroll
  for (int r = 0; r < 8; ++r) { u[r] = v[r] = q[r] = z[r] = cxn[r] = 0.f; }

  int off[8];
#pragma unroll
  for (int r = 0; r < 8; ++r) off[r] = (b0 + r + 8 * laneHi) * HID + n;
#pragma unroll
  for (int r = 0; r < 8; ++r) cx[r] = cur[off[r]];     // t = 0 inputs

  int ycur = y[b0 + wid];                        // uniform per wave -> scalar load
  int ynext = 0;
  float lossacc = 0.f;

  const __bf16* arow = zlds + laneLo * 264;
  __syncthreads();

  for (int t = 0; t < T_STEPS; ++t) {
    // prefetch next step's feedforward current + labels (hides HBM latency)
    if (t + 1 < T_STEPS) {
      size_t base = (size_t)(t + 1) * (BATCH * HID);
#pragma unroll
      for (int r = 0; r < 8; ++r) cxn[r] = cur[base + off[r]];
      ynext = y[(t + 1) * BATCH + b0 + wid];
    }

    // recurrent matmul: z_{t-1} @ W_z^T  (two accumulators break the WMMA RAW chain)
    v8f acc0 = {0.f, 0.f, 0.f, 0.f, 0.f, 0.f, 0.f, 0.f};
    v8f acc1 = {0.f, 0.f, 0.f, 0.f, 0.f, 0.f, 0.f, 0.f};
#pragma unroll
    for (int ks = 0; ks < 8; ks += 2) {
      v16bf a0 = load_a_frag(arow, ks * 32, laneHi);
      v16bf b0f = load_b_frag(wz, HID, n, ks * 32, laneHi);
      v16bf a1 = load_a_frag(arow, (ks + 1) * 32, laneHi);
      v16bf b1f = load_b_frag(wz, HID, n, (ks + 1) * 32, laneHi);
      acc0 = wmma_bf16(a0, b0f, acc0);
      acc1 = wmma_bf16(a1, b1f, acc1);
    }
    __syncthreads();                            // all zlds reads done

    // BRF elementwise (symplectic Euler) + spike + refractory
#pragma unroll
    for (int r = 0; r < 8; ++r) {
      float c = acc0[r] + acc1[r] + cx[r];
      float bb = b0c - q[r];
      u[r] += DELTA_ * (bb * u[r] - omv * v[r] + c);
      v[r] += DELTA_ * (omv * u[r] + bb * v[r]);   // uses updated u
      float zs = (u[r] - THETA_ - q[r]) > 0.f ? 1.f : 0.f;
      q[r] = GAMMA_ * q[r] + zs;
      z[r] = zs;
      zlds[(r + 8 * laneHi) * 264 + n] = (__bf16)zs;
    }
    __syncthreads();                            // z_t visible

    // readout matmul: z_t @ W_o^T (waves 0,1 cover padded O=32)
    if (wid < 2) {
      v8f r0 = {0.f, 0.f, 0.f, 0.f, 0.f, 0.f, 0.f, 0.f};
      v8f r1 = {0.f, 0.f, 0.f, 0.f, 0.f, 0.f, 0.f, 0.f};
#pragma unroll
      for (int ks = 0; ks < 8; ks += 2) {
        v16bf a0 = load_a_frag(arow, ks * 32, laneHi);
        v16bf b0f = load_b_frag(wo, HID, n, ks * 32, laneHi);
        v16bf a1 = load_a_frag(arow, (ks + 1) * 32, laneHi);
        v16bf b1f = load_b_frag(wo, HID, n, (ks + 1) * 32, laneHi);
        r0 = wmma_bf16(a0, b0f, r0);
        r1 = wmma_bf16(a1, b1f, r1);
      }
#pragma unroll
      for (int r = 0; r < 8; ++r) zsl[(r + 8 * laneHi) * 32 + n] = r0[r] + r1[r];
    }
    __syncthreads();                            // zsum ready

    // LI readout update in registers: wave `wid` = row, lane = output channel
    ou = alph * ou + (1.f - alph) * zsl[wid * 32 + lane];   // lanes >=20: alph=0, zsl=0
    if (lane < OUTC)
      d_out[OUTPUTS_OFF + (size_t)t * (BATCH * OUTC) +
            (size_t)(b0 + wid) * OUTC + lane] = ou;

    // wave-parallel log-softmax NLL (shuffle reductions over the 20 valid lanes)
    float mx = (lane < OUTC) ? ou : -3.0e38f;
#pragma unroll
    for (int s = 16; s > 0; s >>= 1) mx = fmaxf(mx, __shfl_xor(mx, s));
    float se = (lane < OUTC) ? __expf(ou - mx) : 0.f;
#pragma unroll
    for (int s = 16; s > 0; s >>= 1) se += __shfl_xor(se, s);
    float ouy = __shfl(ou, ycur);
    lossacc += -(ouy - mx - __logf(se));
    ycur = ynext;

#pragma unroll
    for (int r = 0; r < 8; ++r) cx[r] = cxn[r];
  }

  // final states
#pragma unroll
  for (int r = 0; r < 8; ++r) {
    size_t row = (size_t)(b0 + r + 8 * laneHi);
    d_out[ZF_OFF + row * HID + n] = z[r];
    d_out[UF_OFF + row * HID + n] = u[r];
  }
  if (lane < OUTC)
    d_out[OUF_OFF + (size_t)(b0 + wid) * OUTC + lane] = ou;
  if (lane == 0)
    loss_part[blockIdx.x * 16 + wid] = lossacc;
}

// ---------------- kernel 3: deterministic loss reduction ----------------
__global__ __launch_bounds__(256) void finalize_loss_kernel(
    const float* __restrict__ partials, float* __restrict__ d_out) {
  __shared__ float s[256];
  s[threadIdx.x] = partials[threadIdx.x];
  __syncthreads();
#pragma unroll
  for (int st = 128; st > 0; st >>= 1) {
    if (threadIdx.x < st) s[threadIdx.x] += s[threadIdx.x + st];
    __syncthreads();
  }
  if (threadIdx.x == 0) d_out[LOSS_OFF] = s[0] / (float)BATCH;  // mean over batch
}

// ---------------- host entry ----------------
extern "C" void kernel_launch(void* const* d_in, const int* in_sizes, int n_in,
                              void* d_out, int out_size, void* d_ws, size_t ws_size,
                              hipStream_t stream) {
  (void)in_sizes; (void)n_in; (void)out_size; (void)ws_size;
  const float* x        = (const float*)d_in[0];   // [T,B,I]
  const int*   y        = (const int*)d_in[1];     // [T,B]
  const float* W_h      = (const float*)d_in[2];   // [H, I+H]
  const float* omega    = (const float*)d_in[3];   // [H]
  const float* b_offset = (const float*)d_in[4];   // [H]
  const float* W_o      = (const float*)d_in[5];   // [O, H]
  const float* tau_mem  = (const float*)d_in[6];   // [O]
  float* out = (float*)d_out;

  char* ws = (char*)d_ws;
  __bf16* wx    = (__bf16*)(ws + WS_WX);
  __bf16* wzb   = (__bf16*)(ws + WS_WZ);
  __bf16* wob   = (__bf16*)(ws + WS_WO);
  float*  lpart = (float*)(ws + WS_LOSS);
  float*  cur   = (float*)(ws + WS_CUR);

  prep_weights_kernel<<<(256 * KPAD + 255) / 256, 256, 0, stream>>>(W_h, W_o, wx, wzb, wob);
  gemm_x_kernel<<<(T_STEPS * BATCH) / 16, 512, 0, stream>>>(x, wx, cur);
  brf_scan_kernel<<<BATCH / 16, 512, 0, stream>>>(cur, wzb, wob, y, omega, b_offset,
                                                  tau_mem, out, lpart);
  finalize_loss_kernel<<<1, 256, 0, stream>>>(lpart, out);
}
